// EtcTransientGlobalSelfAttention_73839077753283
// MI455X (gfx1250) — matrix operand order repaired and checked by reading, not compiled, not run on hardware
//
#include <hip/hip_runtime.h>

// ---------------------------------------------------------------------------
// ETC transient-global self-attention for MI455X (gfx1250), bf16 WMMA path.
// B=2, L=4096, D=1024, H=16, dh=64, block_len=128, nb=32, G=256.
// ---------------------------------------------------------------------------

typedef __attribute__((ext_vector_type(16))) __bf16        bf16x16;
typedef __attribute__((ext_vector_type(8)))  float         f32x8;
typedef __attribute__((ext_vector_type(4)))  unsigned int  u32x4;
typedef __attribute__((ext_vector_type(2)))  unsigned int  u32x2;
typedef __attribute__((ext_vector_type(4)))  float         f32x4;
typedef __attribute__((ext_vector_type(4)))  int           i32x4;

union FragB16 { u32x4 u[2]; bf16x16 v; };

// gfx1250 async global->LDS copy (ASYNCcnt path); guarded fallback.
#if defined(__has_builtin)
#if __has_builtin(__builtin_amdgcn_global_load_async_to_lds_b128)
#define HAVE_ASYNC_LDS 1
#endif
#endif
#ifndef HAVE_ASYNC_LDS
#define HAVE_ASYNC_LDS 0
#endif

// Address-space-qualified pointer types matching the builtin signature
// (AS1 prints as __device__, AS3 as __shared__ in HIP diagnostics).
typedef __attribute__((address_space(1))) i32x4 glob_i32x4;
typedef __attribute__((address_space(3))) i32x4 lds_i32x4;

__device__ __forceinline__ unsigned short f2bf(float f) {
  unsigned int u = __float_as_uint(f);
  return (unsigned short)((u + 0x7FFFu + ((u >> 16) & 1u)) >> 16);
}

// ---------------------------------------------------------------------------
// fp32 -> bf16 bulk convert (x rows into Xall[0:8192])
// ---------------------------------------------------------------------------
__global__ __launch_bounds__(256) void k_cvt_bf16(const float* __restrict__ in,
                                                  unsigned short* __restrict__ out,
                                                  int n4) {
  int i = blockIdx.x * 256 + threadIdx.x;
  if (i >= n4) return;
  f32x4 f = ((const f32x4*)in)[i];
  unsigned int lo = (unsigned int)f2bf(f[0]) | ((unsigned int)f2bf(f[1]) << 16);
  unsigned int hi = (unsigned int)f2bf(f[2]) | ((unsigned int)f2bf(f[3]) << 16);
  u32x2 r = {lo, hi};
  ((u32x2*)out)[i] = r;
}

// ---------------------------------------------------------------------------
// transient-global aggregates: g[b,g,:] = sum of 16 token rows of x (fp32 sum)
// written as bf16 rows Xall[8192 + b*256 + g]
// ---------------------------------------------------------------------------
__global__ __launch_bounds__(256) void k_blocksum(const float* __restrict__ x,
                                                  unsigned short* __restrict__ grows) {
  int b = blockIdx.x >> 8;      // 0..1
  int g = blockIdx.x & 255;     // 0..255
  int col = threadIdx.x * 4;    // 0..1023
  f32x4 acc = {0.f, 0.f, 0.f, 0.f};
  const float* base = x + ((size_t)(b * 4096 + g * 16)) * 1024 + col;
#pragma unroll
  for (int r = 0; r < 16; ++r)
    acc += *(const f32x4*)(base + (size_t)r * 1024);
  unsigned int lo = (unsigned int)f2bf(acc[0]) | ((unsigned int)f2bf(acc[1]) << 16);
  unsigned int hi = (unsigned int)f2bf(acc[2]) | ((unsigned int)f2bf(acc[3]) << 16);
  u32x2 r = {lo, hi};
  *(u32x2*)(grows + (size_t)blockIdx.x * 1024 + col) = r;
}

// ---------------------------------------------------------------------------
// weight transpose: Wt[c][d] = W[d][c] (1024x1024), fp32 -> bf16
// ---------------------------------------------------------------------------
__global__ __launch_bounds__(256) void k_transpose_w(const float* __restrict__ w,
                                                     unsigned short* __restrict__ wt) {
  int idx = blockIdx.x * 256 + threadIdx.x;
  int c = idx >> 10, d = idx & 1023;
  wt[(size_t)c * 1024 + d] = f2bf(w[(size_t)d * 1024 + c]);
}

// ---------------------------------------------------------------------------
// Generic bf16 WMMA GEMM: Out[M,1024] = A[M,1024] * Bt[1024,1024]^T (* scale)
// Bt stored N-major: Bt[n][k]. 8 waves per WG as 2(M) x 4(N); wave tile 16x64.
// Register double-buffered so next K-step loads overlap current WMMAs.
// ---------------------------------------------------------------------------
__global__ __launch_bounds__(256) void k_gemm_bf16(const unsigned short* __restrict__ A,
                                                   const unsigned short* __restrict__ Bt,
                                                   void* __restrict__ Out,
                                                   int M, float scale, int out_f32) {
  (void)M;
  const int Kd = 1024, Nd = 1024;
  int wave = threadIdx.x >> 5, lane = threadIdx.x & 31;
  int wm = wave >> 2, wn = wave & 3;
  int half = lane >> 4, lm = lane & 15;
  const int ab = half * 16;                      // A-frag K base (lane group)
  int tileM = blockIdx.x * 32 + wm * 16;
  int tileN = blockIdx.y * 256 + wn * 64;

  const unsigned short* arow  = A  + (size_t)(tileM + lm) * Kd;
  const unsigned short* brow0 = Bt + (size_t)(tileN + lm) * Kd + half * 16;

  f32x8 c[4] = {};
  FragB16 fa[2], fb[2][4];

  auto load_frags = [&](int buf, int kk) {
    fa[buf].u[0] = *(const u32x4*)(arow + kk + ab);        // K = ab+0..7
    fa[buf].u[1] = *(const u32x4*)(arow + kk + ab + 8);    // K = ab+8..15
#pragma unroll
    for (int j = 0; j < 4; ++j) {
      const unsigned short* bp = brow0 + (size_t)j * 16 * Kd + kk;
      fb[buf][j].u[0] = ((const u32x4*)bp)[0];             // K = half*16+0..7
      fb[buf][j].u[1] = ((const u32x4*)bp)[1];             // K = half*16+8..15
    }
  };

  load_frags(0, 0);
#pragma unroll 4
  for (int kk = 0; kk < Kd; kk += 32) {
    int cur = (kk >> 5) & 1;
    if (kk + 32 < Kd) load_frags(cur ^ 1, kk + 32);        // prefetch next step
    __builtin_prefetch(arow + kk + 512, 0, 1);             // global_prefetch_b8
#pragma unroll
    for (int j = 0; j < 4; ++j)
      c[j] = __builtin_amdgcn_wmma_f32_16x16x32_bf16(false, fa[cur].v, false,
                                                     fb[cur][j].v, (short)0,
                                                     c[j], false, false);
  }

#pragma unroll
  for (int j = 0; j < 4; ++j) {
    int col = tileN + j * 16 + lm;
#pragma unroll
    for (int i = 0; i < 8; ++i) {
      int row = tileM + half * 8 + i;
      float v = c[j][i] * scale;
      if (out_f32) ((float*)Out)[(size_t)row * Nd + col] = v;
      else ((unsigned short*)Out)[(size_t)row * Nd + col] = f2bf(v);
    }
  }
}

// ---------------------------------------------------------------------------
// Attention: one WG per (b, block n, head h); 8 waves x 16 queries = 128.
// 640 keys = 3*128 local (masked) + 256 global-side, processed in 20 chunks
// of 32 with flash-style online softmax. The "extra logit" is exact: running
// max init 0, running sum init 1 == phantom logit of 0.
// K chunk is staged cooperatively into LDS via async global->LDS (ASYNCcnt);
// V chunk is staged transposed so P@V B-fragments are contiguous LDS loads.
// ---------------------------------------------------------------------------
__global__ __launch_bounds__(256) void k_attn(const unsigned short* __restrict__ Q,
                                              const unsigned short* __restrict__ Kb,
                                              const unsigned short* __restrict__ Vb,
                                              unsigned short* __restrict__ Y) {
  const int L = 4096, HD = 1024, BL = 128, DH = 64, G = 256, SIDE0 = 8192;
  int h = blockIdx.x & 15;
  int n = (blockIdx.x >> 4) & 31;
  int b = blockIdx.x >> 9;
  int wave = threadIdx.x >> 5, lane = threadIdx.x & 31;
  int half = lane >> 4, lm = lane & 15;
  const int ab = half * 16;

  __shared__ __align__(16) unsigned short s_k[32 * DH];      // K chunk [key][dh]
  __shared__ __align__(16) unsigned short s_vt[DH * 32];     // V^T chunk [dh][key]
  __shared__ __align__(16) unsigned short s_p[8][16 * 32];   // per-wave P tile

  // Q fragments (K-dim = dh): aq0 covers dh 0..31, aq1 covers dh 32..63
  const unsigned short* qp =
      Q + (size_t)(b * L + n * BL + wave * 16 + lm) * HD + h * DH;
  FragB16 aq0, aq1;
  aq0.u[0] = *(const u32x4*)(qp + ab);
  aq0.u[1] = *(const u32x4*)(qp + ab + 8);
  aq1.u[0] = *(const u32x4*)(qp + 32 + ab);
  aq1.u[1] = *(const u32x4*)(qp + 32 + ab + 8);

  f32x8 o[4] = {};
  float rmax[8], rsum[8];
#pragma unroll
  for (int i = 0; i < 8; ++i) { rmax[i] = 0.f; rsum[i] = 1.f; }

  for (int kt = 0; kt < 20; ++kt) {
    __syncthreads();

    // ---- async-stage K chunk [32 keys][64 dh] : 16B per thread ----
    {
      int key = threadIdx.x >> 3;          // 0..31
      int part = threadIdx.x & 7;          // 8 dh elements per part
      int kcol = kt * 32 + key;
      size_t krow;
      if (kcol < 384) {
        int kpos = (n - 1) * BL + kcol;
        if (kpos < 0 || kpos >= L) kpos = 0;   // clamped; bias masks it
        krow = (size_t)(b * L + kpos);
      } else {
        krow = (size_t)(SIDE0 + b * G + (kcol - 384));
      }
      const unsigned short* gp = Kb + krow * HD + h * DH + part * 8;
      unsigned short* lp = &s_k[key * DH + part * 8];
#if HAVE_ASYNC_LDS
      __builtin_amdgcn_global_load_async_to_lds_b128(
          (glob_i32x4*)gp, (lds_i32x4*)lp, 0, 0);
#else
      *(u32x4*)lp = *(const u32x4*)gp;
#endif
    }

    // ---- cooperative V^T staging (64 dh x 32 keys, zero for OOB keys) ----
#pragma unroll
    for (int t = 0; t < 8; ++t) {
      int idx = t * 256 + threadIdx.x;
      int key = idx >> 6;          // 0..31
      int e   = idx & 63;          // dh
      int kcol = kt * 32 + key;
      unsigned short val = 0;
      if (kcol < 384) {
        int kpos = (n - 1) * BL + kcol;
        if (kpos >= 0 && kpos < L)
          val = Vb[(size_t)(b * L + kpos) * HD + h * DH + e];
      } else {
        val = Vb[(size_t)(SIDE0 + b * G + (kcol - 384)) * HD + h * DH + e];
      }
      s_vt[e * 32 + key] = val;
    }
#if HAVE_ASYNC_LDS
    asm volatile("s_wait_asynccnt 0" ::: "memory");
#endif
    __syncthreads();

    // ---- S = Q K^T : two 16x16 tiles over this 32-key chunk (K from LDS) ----
    f32x8 s[2];
#pragma unroll
    for (int t = 0; t < 2; ++t) {
      const unsigned short* kp = &s_k[(t * 16 + lm) * DH + half * 16];
      FragB16 kb0, kb1;
      kb0.u[0] = ((const u32x4*)kp)[0];
      kb0.u[1] = ((const u32x4*)kp)[1];
      kb1.u[0] = ((const u32x4*)(kp + 32))[0];
      kb1.u[1] = ((const u32x4*)(kp + 32))[1];
      f32x8 z = {};
      z = __builtin_amdgcn_wmma_f32_16x16x32_bf16(false, aq0.v, false, kb0.v,
                                                  (short)0, z, false, false);
      z = __builtin_amdgcn_wmma_f32_16x16x32_bf16(false, aq1.v, false, kb1.v,
                                                  (short)0, z, false, false);
      s[t] = z;
    }

    // ---- local mask bias (C layout: elem i -> row half*8+i, lane%16 -> col) --
#pragma unroll
    for (int t = 0; t < 2; ++t) {
      int k_idx = kt * 32 + t * 16 + lm;
      if (k_idx < 384) {
        int kpos = (n - 1) * BL + k_idx;
#pragma unroll
        for (int i = 0; i < 8; ++i) {
          int q_idx = wave * 16 + half * 8 + i;
          int rel = k_idx - BL - q_idx;
          bool valid = (rel >= -127) && (rel <= 127) && (kpos >= 0) && (kpos < L);
          if (!valid) s[t][i] += -1e9f;
        }
      }
    }

    // ---- online softmax update (segmented 16-lane reductions) ----
#pragma unroll
    for (int i = 0; i < 8; ++i) {
      float mx = fmaxf(s[0][i], s[1][i]);
      mx = fmaxf(mx, __shfl_xor(mx, 1, 16));
      mx = fmaxf(mx, __shfl_xor(mx, 2, 16));
      mx = fmaxf(mx, __shfl_xor(mx, 4, 16));
      mx = fmaxf(mx, __shfl_xor(mx, 8, 16));
      float mnew = fmaxf(rmax[i], mx);
      float sc = __expf(rmax[i] - mnew);
      rmax[i] = mnew;
      rsum[i] *= sc;
      o[0][i] *= sc; o[1][i] *= sc; o[2][i] *= sc; o[3][i] *= sc;
      float p0 = __expf(s[0][i] - mnew);
      float p1 = __expf(s[1][i] - mnew);
      float ps = p0 + p1;
      ps += __shfl_xor(ps, 1, 16);
      ps += __shfl_xor(ps, 2, 16);
      ps += __shfl_xor(ps, 4, 16);
      ps += __shfl_xor(ps, 8, 16);
      rsum[i] += ps;
      int m = half * 8 + i;
      s_p[wave][m * 32 + lm]      = f2bf(p0);
      s_p[wave][m * 32 + 16 + lm] = f2bf(p1);
    }

    // intra-wave LDS RAW: P writes -> A-frag reads
    asm volatile("s_wait_dscnt 0" ::: "memory");

    FragB16 pa;   // A-frag of P: row lm, K(keys) chunks {ab..}, {ab+8..}
    const unsigned short* pw = &s_p[wave][0] + lm * 32 + ab;
    pa.u[0] = ((const u32x4*)pw)[0];
    pa.u[1] = ((const u32x4*)(pw + 8))[0];

    // ---- O += P * V  (B-frags contiguous from transposed V in LDS) ----
#pragma unroll
    for (int j = 0; j < 4; ++j) {
      const unsigned short* vp = &s_vt[(j * 16 + lm) * 32 + half * 16];
      FragB16 vb;
      vb.u[0] = ((const u32x4*)vp)[0];
      vb.u[1] = ((const u32x4*)vp)[1];
      o[j] = __builtin_amdgcn_wmma_f32_16x16x32_bf16(false, pa.v, false, vb.v,
                                                     (short)0, o[j], false, false);
    }
  }

  // ---- epilogue: y = O / denom (denom already includes phantom exp(-m)) ----
#pragma unroll
  for (int j = 0; j < 4; ++j) {
    int col = h * DH + j * 16 + lm;
#pragma unroll
    for (int i = 0; i < 8; ++i) {
      int row = b * L + n * BL + wave * 16 + half * 8 + i;
      Y[(size_t)row * HD + col] = f2bf(o[j][i] / rsum[i]);
    }
  }
}

// ---------------------------------------------------------------------------
// Host-side orchestration
// ---------------------------------------------------------------------------
extern "C" void kernel_launch(void* const* d_in, const int* in_sizes, int n_in,
                              void* d_out, int out_size, void* d_ws, size_t ws_size,
                              hipStream_t stream) {
  (void)in_sizes; (void)n_in; (void)out_size; (void)ws_size;
  const float* x  = (const float*)d_in[0];
  const float* Wq = (const float*)d_in[1];
  const float* Wk = (const float*)d_in[2];
  const float* Wv = (const float*)d_in[3];
  const float* Wo = (const float*)d_in[4];

  char* ws = (char*)d_ws;
  // [rows x 1024] bf16 buffers; X/K/V carry 512 extra "global" rows.
  unsigned short* Xall = (unsigned short*)(ws);                 // 8704*1024
  unsigned short* Qb   = (unsigned short*)(ws + 17825792);      // 8192*1024
  unsigned short* Kall = (unsigned short*)(ws + 34603008);      // 8704*1024
  unsigned short* Vall = (unsigned short*)(ws + 52428800);      // 8704*1024
  unsigned short* Yb   = (unsigned short*)(ws + 70254592);      // 8192*1024
  unsigned short* WqT  = (unsigned short*)(ws + 87031808);      // 1024*1024
  unsigned short* WkT  = (unsigned short*)(ws + 89128960);
  unsigned short* WvT  = (unsigned short*)(ws + 91226112);
  unsigned short* WoT  = (unsigned short*)(ws + 93323264);

  // prep: bf16 x, block sums, transposed weights
  k_cvt_bf16<<<8192, 256, 0, stream>>>(x, Xall, 8192 * 1024 / 4);
  k_blocksum<<<512, 256, 0, stream>>>(x, Xall + (size_t)8192 * 1024);
  k_transpose_w<<<4096, 256, 0, stream>>>(Wq, WqT);
  k_transpose_w<<<4096, 256, 0, stream>>>(Wk, WkT);
  k_transpose_w<<<4096, 256, 0, stream>>>(Wv, WvT);
  k_transpose_w<<<4096, 256, 0, stream>>>(Wo, WoT);

  // projections (q scaled by 1/sqrt(dh)=0.125); k/v also over global rows
  k_gemm_bf16<<<dim3(256, 4), 256, 0, stream>>>(Xall, WqT, Qb, 8192, 0.125f, 0);
  k_gemm_bf16<<<dim3(272, 4), 256, 0, stream>>>(Xall, WkT, Kall, 8704, 1.0f, 0);
  k_gemm_bf16<<<dim3(272, 4), 256, 0, stream>>>(Xall, WvT, Vall, 8704, 1.0f, 0);

  // attention: (B=2) * (nb=32) * (H=16) = 1024 workgroups
  k_attn<<<1024, 256, 0, stream>>>(Qb, Kall, Vall, Yb);

  // output projection, fp32 result into d_out
  k_gemm_bf16<<<dim3(256, 4), 256, 0, stream>>>(Yb, WoT, d_out, 8192, 1.0f, 1);
}
